// InteractionMap_29265907155226
// MI455X (gfx1250) — matrix operand order repaired and checked by reading, not compiled, not run on hardware
//
#include <hip/hip_runtime.h>

// ---------------------------------------------------------------------------
// InteractionMap on MI455X (gfx1250, wave32, WMMA)
//   B=1024 graphs, N1=64, N2=128, D=256
//   I      = clip(h1 @ h2^T * D^-0.5, -10, 10)          [64,128]
//   h1_int = LayerNorm(I   @ h2)                        [64,256]
//   h2_int = LayerNorm(I^T @ h1)                        [128,256]
//   out1   = concat(h1, h1_int)  -> [B*64, 512]
//   out2   = concat(h2, h2_int)  -> [B*128, 512]
// Strategy: bf16 WMMA (v_wmma_f32_16x16x32_bf16) with f32 accumulate.
// One workgroup (8 wave32) per graph; all operands resident in LDS (~238KB).
// ---------------------------------------------------------------------------

typedef __attribute__((ext_vector_type(16))) __bf16 v16bf;
typedef __attribute__((ext_vector_type(8)))  float  v8f;

#define BATCH 1024
#define N1 64
#define N2 128
#define DIM 256

// LDS leading dims (elements); each row start stays 16B aligned (ld % 8 == 0)
#define LD_A1  264   // sA1  [64][264]   h1 bf16 row-major
#define LD_A2  264   // sA2  [128][264]  h2 bf16 row-major
#define LD_A1T 72    // sA1T [256][72]   h1^T bf16
#define LD_A2T 136   // sA2T [256][136]  h2^T bf16
#define LD_I   136   // sI   [64][136]   I bf16
#define LD_IT  72    // sIT  [128][72]   I^T bf16

static __device__ __forceinline__ unsigned short f32_to_bf16(float x) {
  unsigned int u = __float_as_uint(x);
  u += 0x7FFFu + ((u >> 16) & 1u);   // round-to-nearest-even
  return (unsigned short)(u >> 16);
}

union FragU { v16bf v; uint4 q[2]; };

// A fragment (16x32 bf16, M x K): lane m (0..15) holds K=[k0..k0+7] and
// K=[k0+16..k0+23]; lane m+16 holds K=[k0+8..15] and K=[k0+24..31].
static __device__ __forceinline__ v16bf load_fragA(const unsigned short* s, int ld,
                                                   int row0, int k0, int lane) {
  const int m = lane & 15, hi = lane >> 4;
  FragU f;
  const unsigned short* p = s + (row0 + m) * ld + k0 + hi * 8;
  f.q[0] = *(const uint4*)(p);
  f.q[1] = *(const uint4*)(p + 16);
  return f.v;
}

// B fragment (32x16 bf16, K x N): lane n (0..15) holds column n, K=[k0..k0+7] and
// K=[k0+16..k0+23]; lane n+16 holds K=[k0+8..15] and K=[k0+24..31]. Source is
// B^T row-major (two 16B LDS loads).
static __device__ __forceinline__ v16bf load_fragB(const unsigned short* sT, int ld,
                                                   int col0, int k0, int lane) {
  const int n = lane & 15, hi = lane >> 4;
  FragU f;
  const unsigned short* p = sT + (col0 + n) * ld + k0 + hi * 8;
  f.q[0] = *(const uint4*)(p);
  f.q[1] = *(const uint4*)(p + 16);
  return f.v;
}

__global__ void __launch_bounds__(256)
interaction_map_kernel(const float* __restrict__ h1g,
                       const float* __restrict__ h2g,
                       float* __restrict__ out) {
  __shared__ __align__(16) unsigned short sA1 [N1  * LD_A1 ];
  __shared__ __align__(16) unsigned short sA2 [N2  * LD_A2 ];
  __shared__ __align__(16) unsigned short sA1T[DIM * LD_A1T];
  __shared__ __align__(16) unsigned short sA2T[DIM * LD_A2T];
  __shared__ __align__(16) unsigned short sI  [N1  * LD_I  ];
  __shared__ __align__(16) unsigned short sIT [N2  * LD_IT ];

  const int b    = blockIdx.x;
  const int tid  = threadIdx.x;
  const int wv   = tid >> 5;
  const int lane = tid & 31;

  float* out1 = out;                                      // [B*64 , 512]
  float* out2 = out + (size_t)BATCH * N1 * 512;           // [B*128, 512]

  // ---- Stage 1: load h1/h2, emit pass-through halves, fill bf16 LDS ----
  {
    const float4* src1 = (const float4*)(h1g + (size_t)b * N1 * DIM);
    for (int idx = tid; idx < N1 * DIM / 4; idx += 256) {
      const int row = idx >> 6;            // DIM/4 = 64 float4 per row
      const int c4  = (idx & 63) * 4;
      float4 v = src1[idx];
      *(float4*)(out1 + ((size_t)(b * N1 + row)) * 512 + c4) = v;
      unsigned short b0 = f32_to_bf16(v.x), b1 = f32_to_bf16(v.y);
      unsigned short b2 = f32_to_bf16(v.z), b3 = f32_to_bf16(v.w);
      uint2 pk; pk.x = b0 | ((unsigned)b1 << 16); pk.y = b2 | ((unsigned)b3 << 16);
      *(uint2*)&sA1[row * LD_A1 + c4] = pk;
      sA1T[(c4 + 0) * LD_A1T + row] = b0;
      sA1T[(c4 + 1) * LD_A1T + row] = b1;
      sA1T[(c4 + 2) * LD_A1T + row] = b2;
      sA1T[(c4 + 3) * LD_A1T + row] = b3;
    }
    const float4* src2 = (const float4*)(h2g + (size_t)b * N2 * DIM);
    for (int idx = tid; idx < N2 * DIM / 4; idx += 256) {
      const int row = idx >> 6;
      const int c4  = (idx & 63) * 4;
      float4 v = src2[idx];
      *(float4*)(out2 + ((size_t)(b * N2 + row)) * 512 + c4) = v;
      unsigned short b0 = f32_to_bf16(v.x), b1 = f32_to_bf16(v.y);
      unsigned short b2 = f32_to_bf16(v.z), b3 = f32_to_bf16(v.w);
      uint2 pk; pk.x = b0 | ((unsigned)b1 << 16); pk.y = b2 | ((unsigned)b3 << 16);
      *(uint2*)&sA2[row * LD_A2 + c4] = pk;
      sA2T[(c4 + 0) * LD_A2T + row] = b0;
      sA2T[(c4 + 1) * LD_A2T + row] = b1;
      sA2T[(c4 + 2) * LD_A2T + row] = b2;
      sA2T[(c4 + 3) * LD_A2T + row] = b3;
    }
  }
  __syncthreads();

  const int n  = lane & 15;
  const int hi = lane >> 4;

  // ---- Stage 2: I = clip(h1 @ h2^T / 16)  (64x128) -> sI, sIT ----
  // wave wv owns column block wv (16 cols), all 4 row blocks.
  for (int ti = 0; ti < 4; ++ti) {
    v8f acc = {0.f, 0.f, 0.f, 0.f, 0.f, 0.f, 0.f, 0.f};
#pragma unroll
    for (int k0 = 0; k0 < DIM; k0 += 32) {
      v16bf a  = load_fragA(sA1, LD_A1, ti * 16, k0, lane);
      v16bf bb = load_fragB(sA2, LD_A2, wv * 16, k0, lane);  // NT: B^T rows = h2 rows
      acc = __builtin_amdgcn_wmma_f32_16x16x32_bf16(false, a, false, bb,
                                                    (short)0, acc, false, false);
    }
    unsigned int pk[4];
#pragma unroll
    for (int v = 0; v < 8; ++v) {
      float x = acc[v] * 0.0625f;                 // D^-0.5
      x = fminf(10.0f, fmaxf(-10.0f, x));
      unsigned short bh = f32_to_bf16(x);
      const int m = ti * 16 + v + 8 * hi;         // C layout: M = v + 8*(lane>=16)
      sI[m * LD_I + wv * 16 + n] = bh;
      if (v & 1) pk[v >> 1] |= ((unsigned int)bh) << 16; else pk[v >> 1] = bh;
    }
    uint4 q; q.x = pk[0]; q.y = pk[1]; q.z = pk[2]; q.w = pk[3];
    *(uint4*)&sIT[(wv * 16 + n) * LD_IT + ti * 16 + hi * 8] = q;  // 8 contiguous M
  }
  __syncthreads();

  // ---- Stage 3a: h1_int_raw = I @ h2  (64x256) -> out1[:,256:512] ----
  for (int t = wv; t < 64; t += 8) {
    const int ti = t >> 4, tj = t & 15;
    v8f acc = {0.f, 0.f, 0.f, 0.f, 0.f, 0.f, 0.f, 0.f};
#pragma unroll
    for (int k0 = 0; k0 < N2; k0 += 32) {
      v16bf a  = load_fragA(sI,   LD_I,   ti * 16, k0, lane);
      v16bf bb = load_fragB(sA2T, LD_A2T, tj * 16, k0, lane);
      acc = __builtin_amdgcn_wmma_f32_16x16x32_bf16(false, a, false, bb,
                                                    (short)0, acc, false, false);
    }
#pragma unroll
    for (int v = 0; v < 8; ++v) {
      const size_t row = (size_t)(b * N1 + ti * 16 + v + 8 * hi);
      out1[row * 512 + 256 + tj * 16 + n] = acc[v];
    }
  }

  // ---- Stage 3b: h2_int_raw = I^T @ h1  (128x256) -> out2[:,256:512] ----
  for (int t = wv; t < 128; t += 8) {
    const int ti = t >> 4, tj = t & 15;
    v8f acc = {0.f, 0.f, 0.f, 0.f, 0.f, 0.f, 0.f, 0.f};
#pragma unroll
    for (int k0 = 0; k0 < N1; k0 += 32) {
      v16bf a  = load_fragA(sIT,  LD_IT,  ti * 16, k0, lane);
      v16bf bb = load_fragB(sA1T, LD_A1T, tj * 16, k0, lane);
      acc = __builtin_amdgcn_wmma_f32_16x16x32_bf16(false, a, false, bb,
                                                    (short)0, acc, false, false);
    }
#pragma unroll
    for (int v = 0; v < 8; ++v) {
      const size_t row = (size_t)(b * N2 + ti * 16 + v + 8 * hi);
      out2[row * 512 + 256 + tj * 16 + n] = acc[v];
    }
  }
}

// ---- LayerNorm (in-place, one wave32 per 256-wide row) ----
__global__ void __launch_bounds__(256)
layernorm_kernel(float* __restrict__ out,
                 const float* __restrict__ gamma,
                 const float* __restrict__ beta) {
  const long R1 = (long)BATCH * N1;
  const long R2 = (long)BATCH * N2;
  const int wv = threadIdx.x >> 5, lane = threadIdx.x & 31;
  long g = (long)blockIdx.x * 8 + wv;
  float* row;
  if (g < R1) {
    row = out + g * 512 + 256;
  } else {
    long g2 = g - R1;
    if (g2 >= R2) return;
    row = out + R1 * 512 + g2 * 512 + 256;
  }
  float x[8], s = 0.f, ss = 0.f;
#pragma unroll
  for (int i = 0; i < 8; ++i) {
    x[i] = row[lane + 32 * i];
    s += x[i];
    ss += x[i] * x[i];
  }
#pragma unroll
  for (int off = 16; off > 0; off >>= 1) {
    s  += __shfl_xor(s,  off, 32);
    ss += __shfl_xor(ss, off, 32);
  }
  const float mu  = s * (1.0f / 256.0f);
  const float var = ss * (1.0f / 256.0f) - mu * mu;
  const float inv = rsqrtf(var + 1e-5f);
#pragma unroll
  for (int i = 0; i < 8; ++i) {
    const int c = lane + 32 * i;
    row[c] = (x[i] - mu) * inv * gamma[c] + beta[c];
  }
}

extern "C" void kernel_launch(void* const* d_in, const int* in_sizes, int n_in,
                              void* d_out, int out_size, void* d_ws, size_t ws_size,
                              hipStream_t stream) {
  const float* h1    = (const float*)d_in[0];
  const float* h2    = (const float*)d_in[1];
  // d_in[2], d_in[3]: batch ids (uniform segments, unused)
  const float* gamma = (const float*)d_in[4];
  const float* beta  = (const float*)d_in[5];
  float* out = (float*)d_out;

  interaction_map_kernel<<<BATCH, 256, 0, stream>>>(h1, h2, out);

  const long rows = (long)BATCH * (N1 + N2);       // 196608 rows, 8 per block
  layernorm_kernel<<<(int)(rows / 8), 256, 0, stream>>>(out, gamma, beta);
}
